// StructureFeatureEncoder_10788957847643
// MI455X (gfx1250) — compile-verified, matrix operand order checked by reading, not run on hardware
//
#include <hip/hip_runtime.h>

#define BB 32
#define NN 256
#define EE 2048
#define DD 256
#define LL 3
#define MM 20
#define KC 256          // K-chunk staged in LDS (== one concat segment)
#define GR 64           // rows per GEMM block (4 row-tiles)

typedef __attribute__((ext_vector_type(16))) __bf16 v16bf;
typedef __attribute__((ext_vector_type(8)))  float  v8f;

// ---------- helpers ----------
__device__ __forceinline__ __bf16 f2bf(float f) {
  union { float f; unsigned u; } in; in.f = f;
  unsigned u = in.u;
  u += 0x7FFFu + ((u >> 16) & 1u);          // round-to-nearest-even
  union { unsigned short s; __bf16 b; } out; out.s = (unsigned short)(u >> 16);
  return out.b;
}
__device__ __forceinline__ unsigned fenc(float f) {
  unsigned u = __float_as_uint(f);
  return (u & 0x80000000u) ? ~u : (u | 0x80000000u);
}
__device__ __forceinline__ float fdec(unsigned u) {
  unsigned v = (u & 0x80000000u) ? (u & 0x7FFFFFFFu) : ~u;
  return __uint_as_float(v);
}
__device__ __forceinline__ unsigned pcg(unsigned x) {
  x = x * 747796405u + 2891336453u;
  unsigned w = ((x >> ((x >> 28) + 4u)) ^ x) * 277803737u;
  return (w >> 22) ^ w;
}

// 16-bit WMMA fragment mapping: K offset-in-32 `co` -> (half, elem)
//   lanes 0-15  hold K 0-7 (e0-7)  and 16-23 (e8-15)
//   lanes 16-31 hold K 8-15 (e0-7) and 24-31 (e8-15)
__device__ __forceinline__ int frag_half(int co) { return (co >> 3) & 1; }
__device__ __forceinline__ int frag_elem(int co) { return (co & 7) + ((co >> 4) << 3); }

// LDS A slice: [rt][ksl][lane][elem], rt=row/16 (4), ksl=K/32 within chunk (8)
__device__ __forceinline__ void sA_store_chunk(__bf16* sA, int r, int c, float v) {
  const int rt = r >> 4, rr = r & 15;
  const int ksl = c >> 5, co = c & 31;
  const int lane = frag_half(co) * 16 + rr;
  sA[(((rt * (KC / 32) + ksl) * 32) + lane) * 16 + frag_elem(co)] = f2bf(v);
}

// one K-chunk of wave-level compute: acc[rt][t] += A[rt] * B[t]
// Wsh fragment order: (((tc*KS + ks)*32 + lane)*16 + elem), tc = col/16
template<int KS_TOTAL>
__device__ __forceinline__ void wmma_chunk(const __bf16* sA, const __bf16* Wsh,
                                           v8f acc[4][2], int wave, int lane, int ks0) {
#pragma unroll
  for (int ksl = 0; ksl < KC / 32; ++ksl) {
    const int ks = ks0 + ksl;
    v16bf a[4];
#pragma unroll
    for (int rt = 0; rt < 4; ++rt)
      a[rt] = *(const v16bf*)(sA + ((size_t)(rt * (KC / 32) + ksl) * 32 + lane) * 16);
#pragma unroll
    for (int t = 0; t < 2; ++t) {
      const int tc = wave * 2 + t;
      v16bf b = *(const v16bf*)(Wsh + (((size_t)tc * KS_TOTAL + ks) * 32 + lane) * 16);
#pragma unroll
      for (int rt = 0; rt < 4; ++rt)
        acc[rt][t] = __builtin_amdgcn_wmma_f32_16x16x32_bf16(false, a[rt], false, b,
                                                             (short)0, acc[rt][t],
                                                             false, false);
    }
  }
}

// ---------- weight shuffle: f32 (KK x 256) -> bf16 fragment order ----------
__global__ __launch_bounds__(256)
void shuffle_w_kernel(const float* __restrict__ W, __bf16* __restrict__ dst, int KK) {
  const int idx = blockIdx.x * 256 + threadIdx.x;       // over KK*DD
  const int kk = idx / DD, col = idx % DD;
  const int tc = col >> 4, lrow = col & 15;
  const int ks = kk >> 5, co = kk & 31;
  const int lane = frag_half(co) * 16 + lrow;
  const int KS = KK >> 5;
  dst[(((size_t)tc * KS + ks) * 32 + lane) * 16 + frag_elem(co)] = f2bf(W[idx]);
}

// ---------- h = dist_embed[clip(dists)] + 0.1*noise ; keep dist_emb ----------
__global__ __launch_bounds__(256)
void node_embed_kernel(const int* __restrict__ dists, const float* __restrict__ dist_embed,
                       float* __restrict__ h, float* __restrict__ de) {
  const int idx = blockIdx.x * 256 + threadIdx.x;       // over B*N*D
  const int nd = idx / DD;
  const int d  = idx % DD;
  int q = dists[nd]; q = q < 0 ? 0 : (q > 9 ? 9 : q);
  float emb = dist_embed[q * DD + d];
  unsigned h1 = pcg(2u * (unsigned)idx + 1u);
  unsigned h2 = pcg(2u * (unsigned)idx + 2u);
  float u1 = ((float)h1 + 0.5f) * 2.3283064e-10f;
  float u2 = ((float)h2 + 0.5f) * 2.3283064e-10f;
  float z = sqrtf(-2.0f * __logf(u1)) * __cosf(6.2831853f * u2);
  de[idx] = emb;
  h[idx]  = emb + 0.1f * z;
}

// ---------- per-layer init: aggr=0, sm=0, mx=enc(-1e9) ----------
__global__ __launch_bounds__(256)
void init_layer_kernel(float* __restrict__ aggr, float* __restrict__ sm,
                       unsigned* __restrict__ mxe) {
  int i = blockIdx.x * 256 + threadIdx.x;               // over B*N*D
  aggr[i] = 0.f;
  if (i < BB * NN) { sm[i] = 0.f; mxe[i] = fenc(-1e9f); }
}

// ---------- den = sigmoid(relu([h_r|r_q|conf] @ W1 + b1) @ W2 + b2) * edge_mask ----------
__global__ __launch_bounds__(256)
void den_gemm_kernel(const int* __restrict__ rels, const float* __restrict__ rel_embed,
                     const float* __restrict__ rq, const float* __restrict__ conf,
                     const __bf16* __restrict__ W1sh, const float* __restrict__ b1,
                     const float* __restrict__ W2, const float* __restrict__ b2,
                     const unsigned char* __restrict__ emask, float* __restrict__ den) {
  __shared__ alignas(32) __bf16 sA[GR * KC];            // 32 KB
  __shared__ float sRow[GR];
  __shared__ int sB[GR], sRel[GR];
  const int tid = threadIdx.x;
  const int rowBase = blockIdx.x * GR;                  // edge rows (b*E+e)
  if (tid < GR) {
    int ge = rowBase + tid;
    sB[tid] = ge / EE;
    sRel[tid] = rels[ge];
    sRow[tid] = 0.f;
  }
  const int wave = tid >> 5, lane = tid & 31;
  v8f acc[4][2] = {};
  for (int seg = 0; seg < 3; ++seg) {
    __syncthreads();
    for (int idx = tid; idx < GR * KC; idx += 256) {
      int r = idx >> 8, d = idx & 255;
      float v;
      if (seg == 0)      v = rel_embed[sRel[r] * DD + d];
      else if (seg == 1) v = rq[sB[r] * DD + d];
      else               v = conf[((size_t)rowBase + r) * DD + d];
      sA_store_chunk(sA, r, d, v);
    }
    __syncthreads();
    wmma_chunk<24>(sA, W1sh, acc, wave, lane, seg * (KC / 32));
  }
  const int half = lane >> 4, lrow = lane & 15;
#pragma unroll
  for (int rt = 0; rt < 4; ++rt) {
    float part[8];
#pragma unroll
    for (int r = 0; r < 8; ++r) part[r] = 0.f;
#pragma unroll
    for (int t = 0; t < 2; ++t) {
      const int col = wave * 32 + t * 16 + lrow;
      const float bias = b1[col], w2 = W2[col];
#pragma unroll
      for (int r = 0; r < 8; ++r) {
        float v = acc[rt][t][r] + bias;
        part[r] += (v > 0.f ? v : 0.f) * w2;
      }
    }
#pragma unroll
    for (int r = 0; r < 8; ++r) atomicAdd(&sRow[rt * 16 + r + half * 8], part[r]);
  }
  __syncthreads();
  if (tid < GR) {
    int ge = rowBase + tid;
    float x = sRow[tid] + b2[0];
    float sg = 1.f / (1.f + __expf(-x));
    den[ge] = sg * (emask[ge] ? 1.f : 0.f);
  }
}

// ---------- msg = relu([h_src*h_r | h_src | dist_src | h_r | conf] @ msg_W[k] + b) ----------
__global__ __launch_bounds__(256)
void msg_gemm_kernel(int k, const float* __restrict__ h, const float* __restrict__ de,
                     const int* __restrict__ edge_index, const int* __restrict__ rels,
                     const float* __restrict__ rel_embed, const float* __restrict__ conf,
                     const __bf16* __restrict__ msgWsh, const float* __restrict__ msg_b,
                     float* __restrict__ msg) {
  __shared__ alignas(32) __bf16 sA[GR * KC];            // 32 KB
  __shared__ int sB[GR], sE[GR], sSrc[GR], sRel[GR];
  const int tid = threadIdx.x;
  const int rowBase = blockIdx.x * GR;
  if (tid < GR) {
    int ge = rowBase + tid;
    int b = ge / EE, e = ge % EE;
    sB[tid] = b; sE[tid] = e;
    sSrc[tid] = edge_index[b * 2 * EE + e];             // src node
    sRel[tid] = rels[b * EE + e];
  }
  const int wave = tid >> 5, lane = tid & 31;
  const __bf16* Wsh = msgWsh + (size_t)k * 5 * DD * DD;
  v8f acc[4][2] = {};
  for (int seg = 0; seg < 5; ++seg) {
    __syncthreads();
    for (int idx = tid; idx < GR * KC; idx += 256) {
      int r = idx >> 8, d = idx & 255;
      int b = sB[r], e = sE[r], src = sSrc[r];
      float v;
      if (seg == 0)      v = h[((size_t)b * NN + src) * DD + d] * rel_embed[sRel[r] * DD + d];
      else if (seg == 1) v = h[((size_t)b * NN + src) * DD + d];
      else if (seg == 2) v = de[((size_t)b * NN + src) * DD + d];
      else if (seg == 3) v = rel_embed[sRel[r] * DD + d];
      else               v = conf[((size_t)b * EE + e) * DD + d];
      sA_store_chunk(sA, r, d, v);
    }
    __syncthreads();
    wmma_chunk<40>(sA, Wsh, acc, wave, lane, seg * (KC / 32));
  }
  const int half = lane >> 4, lrow = lane & 15;
  const float* bias = msg_b + k * DD;
#pragma unroll
  for (int rt = 0; rt < 4; ++rt)
#pragma unroll
    for (int t = 0; t < 2; ++t) {
      const int col = wave * 32 + t * 16 + lrow;
#pragma unroll
      for (int r = 0; r < 8; ++r) {
        const int row = rowBase + rt * 16 + r + half * 8;
        float v = acc[rt][t][r] + bias[col];
        msg[(size_t)row * DD + col] = v > 0.f ? v : 0.f;
      }
    }
}

// ---------- score = leaky_relu([msg|h_r|r_q] @ attL_W[k] + b); scatter-max ----------
__global__ __launch_bounds__(256)
void score_kernel(int k, const float* __restrict__ msg, const int* __restrict__ rels,
                  const float* __restrict__ rel_embed, const float* __restrict__ rq,
                  const int* __restrict__ edge_index, const unsigned char* __restrict__ emask,
                  const float* __restrict__ attLW, const float* __restrict__ attLb,
                  float* __restrict__ score, unsigned* __restrict__ mxe) {
  const int wave = threadIdx.x >> 5, lane = threadIdx.x & 31;
  const int ge = blockIdx.x * 8 + wave;
  const int b = ge / EE, e = ge % EE;
  const float* wv = attLW + k * 3 * DD;
  const int rel = rels[b * EE + e];
  float s = 0.f;
  for (int d = lane; d < DD; d += 32) {
    s += msg[(size_t)ge * DD + d] * wv[d];
    s += rel_embed[rel * DD + d] * wv[DD + d];
    s += rq[b * DD + d] * wv[2 * DD + d];
  }
  for (int o = 16; o; o >>= 1) s += __shfl_down(s, o, 32);
  if (lane == 0) {
    s += attLb[k];
    s = s > 0.f ? s : 0.01f * s;
    if (!emask[b * EE + e]) s = -1e9f;
    score[ge] = s;
    const int tgt = edge_index[b * 2 * EE + EE + e];
    atomicMax(&mxe[b * NN + tgt], fenc(s));
  }
}

// ---------- ex = exp(score - mx[tgt]); sm[tgt] += ex ----------
__global__ __launch_bounds__(256)
void exsm_kernel(const float* __restrict__ score, const int* __restrict__ edge_index,
                 const unsigned* __restrict__ mxe, float* __restrict__ exv,
                 float* __restrict__ sm) {
  const int ge = blockIdx.x * 256 + threadIdx.x;
  const int b = ge / EE, e = ge % EE;
  const int tgt = edge_index[b * 2 * EE + EE + e];
  const float ex = __expf(score[ge] - fdec(mxe[b * NN + tgt]));
  exv[ge] = ex;
  atomicAdd(&sm[b * NN + tgt], ex);
}

// ---------- aggr[tgt] += alpha * msg * em * den ----------
__global__ __launch_bounds__(256)
void aggr_kernel(const float* __restrict__ msg, const float* __restrict__ exv,
                 const float* __restrict__ sm, const float* __restrict__ den,
                 const unsigned char* __restrict__ emask, const int* __restrict__ edge_index,
                 float* __restrict__ aggr) {
  const int ge = blockIdx.x;
  const int b = ge / EE, e = ge % EE;
  const int tgt = edge_index[b * 2 * EE + EE + e];
  const float coef = exv[ge] / (sm[b * NN + tgt] + 1e-8f) * den[ge]
                   * (emask[b * EE + e] ? 1.f : 0.f);
  const int d = threadIdx.x;
  atomicAdd(&aggr[((size_t)b * NN + tgt) * DD + d], coef * msg[(size_t)ge * DD + d]);
}

// ---------- h = aggr @ upd_W[k] + upd_b[k] + h ; outs[k] = h ----------
__global__ __launch_bounds__(256)
void upd_gemm_kernel(int k, const float* __restrict__ aggr, const __bf16* __restrict__ updWsh,
                     const float* __restrict__ updb, float* __restrict__ h,
                     float* __restrict__ outs) {
  __shared__ alignas(32) __bf16 sA[GR * KC];            // 32 KB
  const int tid = threadIdx.x;
  const int rowBase = blockIdx.x * GR;                  // node rows (b*N+n)
  for (int idx = tid; idx < GR * KC; idx += 256) {
    int r = idx >> 8, d = idx & 255;
    sA_store_chunk(sA, r, d, aggr[(size_t)rowBase * DD + idx]);
  }
  __syncthreads();
  const int wave = tid >> 5, lane = tid & 31;
  v8f acc[4][2] = {};
  wmma_chunk<8>(sA, updWsh + (size_t)k * DD * DD, acc, wave, lane, 0);
  const int half = lane >> 4, lrow = lane & 15;
  const float* bias = updb + k * DD;
  float* ok = outs + (size_t)k * BB * NN * DD;
#pragma unroll
  for (int rt = 0; rt < 4; ++rt)
#pragma unroll
    for (int t = 0; t < 2; ++t) {
      const int col = wave * 32 + t * 16 + lrow;
#pragma unroll
      for (int r = 0; r < 8; ++r) {
        const int row = rowBase + rt * 16 + r + half * 8;
        float v = acc[rt][t][r] + bias[col] + h[(size_t)row * DD + col];
        h[(size_t)row * DD + col] = v;
        ok[(size_t)row * DD + col] = v;
      }
    }
}

// ---------- JK: softmax over layer scores, weighted sum, mask ----------
__global__ __launch_bounds__(256)
void jk_kernel(const float* __restrict__ outs, const float* __restrict__ jkW,
               const float* __restrict__ jkb, const unsigned char* __restrict__ mask,
               float* __restrict__ h) {
  const int tid = threadIdx.x, wave = tid >> 5, lane = tid & 31;
  const int row = blockIdx.x * 8 + wave;                // 0..B*N-1
  const size_t stride = (size_t)BB * NN * DD;
  float s[LL];
#pragma unroll
  for (int k = 0; k < LL; ++k) {
    float p = 0.f;
    for (int d = lane; d < DD; d += 32)
      p += outs[(size_t)k * stride + (size_t)row * DD + d] * jkW[d];
    for (int o = 16; o; o >>= 1) p += __shfl_down(p, o, 32);
    p = __shfl(p, 0, 32);
    s[k] = p + jkb[0];
  }
  float mx = fmaxf(s[0], fmaxf(s[1], s[2]));
  float e0 = __expf(s[0] - mx), e1 = __expf(s[1] - mx), e2 = __expf(s[2] - mx);
  float inv = 1.f / (e0 + e1 + e2);
  float w0 = e0 * inv, w1 = e1 * inv, w2 = e2 * inv;
  float mk = mask[row] ? 1.f : 0.f;
  for (int d = lane; d < DD; d += 32) {
    size_t o0 = (size_t)row * DD + d;
    h[o0] = (w0 * outs[o0] + w1 * outs[stride + o0] + w2 * outs[2 * stride + o0]) * mk;
  }
}

// ---------- final attention + softmax + top-20 + outputs ----------
__global__ __launch_bounds__(256)
void final_kernel(const float* __restrict__ h, const float* __restrict__ rq,
                  const unsigned char* __restrict__ mask,
                  const float* __restrict__ attW, const float* __restrict__ attb,
                  float* __restrict__ out) {
  __shared__ float sAl[NN];
  __shared__ float rv[NN];
  __shared__ int   ri[NN];
  __shared__ float stv[MM];
  __shared__ int   sti[MM];
  const int b = blockIdx.x, n = threadIdx.x;
  float s = 0.f;
  for (int d = 0; d < DD; ++d) {
    s += h[((size_t)b * NN + n) * DD + d] * attW[d];
    s += rq[b * DD + d] * attW[DD + d];
  }
  s += attb[0];
  s = s > 0.f ? s : 0.01f * s;
  if (!mask[b * NN + n]) s = -1e9f;
  rv[n] = s; __syncthreads();
  for (int st = 128; st; st >>= 1) { if (n < st) rv[n] = fmaxf(rv[n], rv[n + st]); __syncthreads(); }
  const float mx = rv[0]; __syncthreads();
  const float ex = __expf(s - mx);
  rv[n] = ex; __syncthreads();
  for (int st = 128; st; st >>= 1) { if (n < st) rv[n] += rv[n + st]; __syncthreads(); }
  const float sum = rv[0]; __syncthreads();
  sAl[n] = ex / sum; __syncthreads();
  for (int m = 0; m < MM; ++m) {
    rv[n] = sAl[n]; ri[n] = n; __syncthreads();
    for (int st = 128; st; st >>= 1) {
      if (n < st) {
        if (rv[n + st] > rv[n] || (rv[n + st] == rv[n] && ri[n + st] < ri[n])) {
          rv[n] = rv[n + st]; ri[n] = ri[n + st];
        }
      }
      __syncthreads();
    }
    if (n == 0) { stv[m] = rv[0]; sti[m] = ri[0]; sAl[ri[0]] = -1.f; }
    __syncthreads();
  }
  for (int m = 0; m < MM; ++m)
    out[((size_t)b * MM + m) * DD + n] = h[((size_t)b * NN + sti[m]) * DD + n] * stv[m];
  out[(size_t)BB * MM * DD + (size_t)b * DD + n] = h[(size_t)b * NN * DD + n]; // t_state
}

// ---------- host ----------
extern "C" void kernel_launch(void* const* d_in, const int* in_sizes, int n_in,
                              void* d_out, int out_size, void* d_ws, size_t ws_size,
                              hipStream_t stream) {
  (void)in_sizes; (void)n_in; (void)out_size; (void)ws_size;
  const int* dists        = (const int*)d_in[0];
  const int* edge_index   = (const int*)d_in[1];
  const int* rels         = (const int*)d_in[2];
  const unsigned char* mask  = (const unsigned char*)d_in[3];
  const unsigned char* emask = (const unsigned char*)d_in[4];
  const float* rq         = (const float*)d_in[5];
  const float* conf       = (const float*)d_in[6];
  const float* rel_embed  = (const float*)d_in[7];
  const float* dist_embed = (const float*)d_in[8];
  const float* msg_W      = (const float*)d_in[9];
  const float* msg_b      = (const float*)d_in[10];
  const float* upd_W      = (const float*)d_in[11];
  const float* upd_b      = (const float*)d_in[12];
  const float* attL_W     = (const float*)d_in[13];
  const float* attL_b     = (const float*)d_in[14];
  const float* att_W      = (const float*)d_in[15];
  const float* att_b      = (const float*)d_in[16];
  const float* jk_W       = (const float*)d_in[17];
  const float* jk_b       = (const float*)d_in[18];
  const float* den_W1     = (const float*)d_in[19];
  const float* den_b1     = (const float*)d_in[20];
  const float* den_W2     = (const float*)d_in[21];
  const float* den_b2     = (const float*)d_in[22];

  const size_t ND = (size_t)BB * NN * DD;       // 2,097,152
  const size_t ED = (size_t)BB * EE * DD;       // 16,777,216
  float* h    = (float*)d_ws;
  float* de   = h + ND;
  float* aggr = de + ND;
  float* outs = aggr + ND;                      // 3*ND
  float* msg  = outs + 3 * ND;
  float* den  = msg + ED;                       // B*E
  float* score= den + BB * EE;
  float* exv  = score + BB * EE;
  float* sm   = exv + BB * EE;                  // B*N
  unsigned* mxe = (unsigned*)(sm + BB * NN);    // B*N
  __bf16* msgWsh = (__bf16*)(mxe + BB * NN);    // 3*1280*256 (32B-aligned offset)
  __bf16* denW1sh = msgWsh + (size_t)LL * 5 * DD * DD;
  __bf16* updWsh  = denW1sh + (size_t)3 * DD * DD;

  // shuffle weights into WMMA fragment order (bf16)
  for (int k = 0; k < LL; ++k) {
    shuffle_w_kernel<<<(5 * DD * DD) / 256, 256, 0, stream>>>(
        msg_W + (size_t)k * 5 * DD * DD, msgWsh + (size_t)k * 5 * DD * DD, 5 * DD);
    shuffle_w_kernel<<<(DD * DD) / 256, 256, 0, stream>>>(
        upd_W + (size_t)k * DD * DD, updWsh + (size_t)k * DD * DD, DD);
  }
  shuffle_w_kernel<<<(3 * DD * DD) / 256, 256, 0, stream>>>(den_W1, denW1sh, 3 * DD);

  node_embed_kernel<<<(int)(ND / 256), 256, 0, stream>>>(dists, dist_embed, h, de);

  den_gemm_kernel<<<BB * EE / GR, 256, 0, stream>>>(rels, rel_embed, rq, conf,
                                                    denW1sh, den_b1, den_W2, den_b2,
                                                    emask, den);
  for (int k = 0; k < LL; ++k) {
    init_layer_kernel<<<(int)(ND / 256), 256, 0, stream>>>(aggr, sm, mxe);
    msg_gemm_kernel<<<BB * EE / GR, 256, 0, stream>>>(k, h, de, edge_index, rels,
                                                      rel_embed, conf, msgWsh, msg_b, msg);
    score_kernel<<<BB * EE / 8, 256, 0, stream>>>(k, msg, rels, rel_embed, rq, edge_index,
                                                  emask, attL_W, attL_b, score, mxe);
    exsm_kernel<<<BB * EE / 256, 256, 0, stream>>>(score, edge_index, mxe, exv, sm);
    aggr_kernel<<<BB * EE, 256, 0, stream>>>(msg, exv, sm, den, emask, edge_index, aggr);
    upd_gemm_kernel<<<BB * NN / GR, 256, 0, stream>>>(k, aggr, updWsh, upd_b, h, outs);
  }
  jk_kernel<<<BB * NN / 8, 256, 0, stream>>>(outs, jk_W, jk_b, mask, h);
  final_kernel<<<BB, 256, 0, stream>>>(h, rq, mask, att_W, att_b, (float*)d_out);
}
